// MultiHeadedAttention_34729105555695
// MI455X (gfx1250) — compile-verified, matrix operand order checked by reading, not run on hardware
//
#include <hip/hip_runtime.h>
#include <hip/hip_bf16.h>

// ---------------- types ----------------
typedef __attribute__((ext_vector_type(16))) __bf16          v16bf;
typedef __attribute__((ext_vector_type(16))) unsigned short  v16us;
typedef __attribute__((ext_vector_type(8)))  unsigned short  v8us;
typedef __attribute__((ext_vector_type(4)))  unsigned short  v4us;
typedef __attribute__((ext_vector_type(8)))  float           v8f;
typedef __attribute__((ext_vector_type(4)))  int             v4i;

#define S_LEN 2048
#define D_MODEL 1024
#define NHEAD 16
#define DK 64
#define BATCH 2

// ---------------- gfx1250 async global->LDS path (guarded) ----------------
#if defined(__has_builtin)
#if __has_builtin(__builtin_amdgcn_global_load_async_to_lds_b128) && \
    __has_builtin(__builtin_amdgcn_s_wait_asynccnt)
#define USE_ASYNC_LDS 1
#endif
#endif
#ifndef USE_ASYNC_LDS
#define USE_ASYNC_LDS 0
#endif

__device__ __forceinline__ unsigned short f32_to_bf16_rne(float f) {
    unsigned u = __builtin_bit_cast(unsigned, f);
    unsigned r = u + 0x7FFFu + ((u >> 16) & 1u);
    return (unsigned short)(r >> 16);
}

// copy 16 bytes global -> LDS (async on gfx1250)
__device__ __forceinline__ void copy16_g2l(const unsigned short* g, unsigned short* l) {
#if USE_ASYNC_LDS
    __builtin_amdgcn_global_load_async_to_lds_b128((v4i*)(void*)g, (v4i*)(void*)l, 0, 0);
#else
    *(v8us*)l = *(const v8us*)g;
#endif
}

// Build a 16x32 bf16 A/B fragment for v_wmma_f32_16x16x32_bf16.
// p points at (row base + khalf); elements {0..7} = p[0..7], {8..15} = p[16..23].
__device__ __forceinline__ v16bf load_frag(const unsigned short* p) {
    v8us lo = *(const v8us*)(p);
    v8us hi = *(const v8us*)(p + 16);
    v16us t;
#pragma unroll
    for (int i = 0; i < 8; ++i) { t[i] = lo[i]; t[i + 8] = hi[i]; }
    return __builtin_bit_cast(v16bf, t);
}

__device__ __forceinline__ v8f wmma_bf16(v16bf a, v16bf b, v8f c) {
    return __builtin_amdgcn_wmma_f32_16x16x32_bf16(
        /*neg_a=*/false, a, /*neg_b=*/false, b,
        /*c_mod=*/(short)0, c, /*reuse_a=*/false, /*reuse_b=*/false);
}

// ---------------- f32 -> bf16 conversion ----------------
__global__ void cvt4_kernel(const float* __restrict__ in,
                            unsigned short* __restrict__ out, int n4) {
    int i = blockIdx.x * blockDim.x + threadIdx.x;
    if (i < n4) {
        float4 f = ((const float4*)in)[i];
        v4us o;
        o[0] = f32_to_bf16_rne(f.x);
        o[1] = f32_to_bf16_rne(f.y);
        o[2] = f32_to_bf16_rne(f.z);
        o[3] = f32_to_bf16_rne(f.w);
        ((v4us*)out)[i] = o;
    }
}

// ---------------- LDS-tiled GEMM: Y[m][n] = sum_k X[m][k]*W[n][k] + bias[n] ----
// Block = 256 threads (8 waves) computing a 128x128 output tile.
// A/B staged in double-buffered 128x32 bf16 LDS tiles via async global->LDS,
// overlapped with WMMA compute (s_wait_asynccnt pipelining).
// MODE 0: out bf16 head-split  [B][H][S][DK]
// MODE 1: out bf16 transposed  [B][H][DK][S]   (for V)
// MODE 2: out f32              [4096][1024]    (final projection)
template <int MODE>
__global__ void gemm_bf16_kernel(const unsigned short* __restrict__ X,
                                 const unsigned short* __restrict__ W,
                                 const float* __restrict__ bias,
                                 void* __restrict__ out) {
    __shared__ unsigned short At[2][128 * 32];
    __shared__ unsigned short Bt[2][128 * 32];

    int tid   = threadIdx.x;
    int lane  = tid & 31;
    int wave  = tid >> 5;
    int ln    = lane & 15;
    int khalf = (lane >> 4) << 3;     // 0 or 8
    int mbase = (lane >> 4) << 3;
    int wm    = wave >> 1;            // 0..3 : 32-row slice
    int wn    = wave & 1;             // 0..1 : 64-col slice
    int m0    = blockIdx.y << 7;
    int n0    = blockIdx.x << 7;

    // tile staging: 128x32 u16 = 8KB = 512 x 16B; 256 threads -> 2 chunks each
    int r0 = tid >> 2,          c0c = (tid & 3) << 3;
    int r1 = (tid + 256) >> 2,  c1c = c0c;           // (idx&3) identical for +256

    auto load_tiles = [&](int buf, int k0) {
        copy16_g2l(X + (size_t)(m0 + r0) * D_MODEL + k0 + c0c, &At[buf][r0 * 32 + c0c]);
        copy16_g2l(X + (size_t)(m0 + r1) * D_MODEL + k0 + c1c, &At[buf][r1 * 32 + c1c]);
        copy16_g2l(W + (size_t)(n0 + r0) * D_MODEL + k0 + c0c, &Bt[buf][r0 * 32 + c0c]);
        copy16_g2l(W + (size_t)(n0 + r1) * D_MODEL + k0 + c1c, &Bt[buf][r1 * 32 + c1c]);
    };

    v8f acc[2][4] = {};

    load_tiles(0, 0);
    for (int k0 = 0; k0 < D_MODEL; k0 += 32) {
        int buf = (k0 >> 5) & 1;
#if USE_ASYNC_LDS
        if (k0 + 32 < D_MODEL) {
            load_tiles(buf ^ 1, k0 + 32);
            __builtin_amdgcn_s_wait_asynccnt(4);  // in-order: current tile's 4 done
        } else {
            __builtin_amdgcn_s_wait_asynccnt(0);
        }
#else
        if (k0 + 32 < D_MODEL) load_tiles(buf ^ 1, k0 + 32);
#endif
        __syncthreads();

        v16bf af[2], bfr[4];
#pragma unroll
        for (int mt = 0; mt < 2; ++mt)
            af[mt] = load_frag(&At[buf][(wm * 32 + mt * 16 + ln) * 32 + khalf]);
#pragma unroll
        for (int nt = 0; nt < 4; ++nt)
            bfr[nt] = load_frag(&Bt[buf][(wn * 64 + nt * 16 + ln) * 32 + khalf]);
#pragma unroll
        for (int mt = 0; mt < 2; ++mt)
#pragma unroll
            for (int nt = 0; nt < 4; ++nt)
                acc[mt][nt] = wmma_bf16(af[mt], bfr[nt], acc[mt][nt]);
        __syncthreads();
    }

    // epilogue: bias + store per MODE
#pragma unroll
    for (int mt = 0; mt < 2; ++mt) {
#pragma unroll
        for (int nt = 0; nt < 4; ++nt) {
            int col = n0 + wn * 64 + nt * 16 + ln;
            float bval = bias[col];
#pragma unroll
            for (int r = 0; r < 8; ++r) {
                int row = m0 + wm * 32 + mt * 16 + mbase + r;
                float v = acc[mt][nt][r] + bval;
                if (MODE == 2) {
                    ((float*)out)[(size_t)row * D_MODEL + col] = v;
                } else {
                    int b = row >> 11, s = row & (S_LEN - 1);
                    int h = col >> 6, d = col & (DK - 1);
                    unsigned short hv = f32_to_bf16_rne(v);
                    if (MODE == 0)
                        ((unsigned short*)out)[(((size_t)(b * NHEAD + h) * S_LEN) + s) * DK + d] = hv;
                    else
                        ((unsigned short*)out)[(((size_t)(b * NHEAD + h) * DK) + d) * S_LEN + s] = hv;
                }
            }
        }
    }
}

// ---------------- Flash-style attention, one wave per 16-row query tile ----------------
// Qh,Kh: [B][H][S][DK] bf16.  Vt: [B][H][DK][S] bf16.  mask: [B][S][S] int.
// ctx out: [B][S][D_MODEL] bf16 (heads concatenated).
__global__ void attn_kernel(const unsigned short* __restrict__ Qh,
                            const unsigned short* __restrict__ Kh,
                            const unsigned short* __restrict__ Vt,
                            const int* __restrict__ mask,
                            unsigned short* __restrict__ ctx) {
    __shared__ unsigned short ldsP[8][16 * 32];   // per-wave P tile (16 q-rows x 32 keys)

    int lane  = threadIdx.x & 31;
    int wave  = threadIdx.x >> 5;
    int ln    = lane & 15;
    int khalf = (lane >> 4) << 3;
    int mbase = (lane >> 4) << 3;

    int wid   = blockIdx.x * 8 + wave;       // 0..4095
    int bh    = wid >> 7;                    // tile count per (b,h) = S/16 = 128
    int stile = wid & 127;
    int b     = bh >> 4;
    int h     = bh & 15;
    int q0    = stile << 4;

    const unsigned short* Qb = Qh + (size_t)bh * S_LEN * DK;
    const unsigned short* Kb = Kh + (size_t)bh * S_LEN * DK;
    const unsigned short* Vb = Vt + (size_t)bh * DK * S_LEN;
    const int* mb            = mask + (size_t)b * S_LEN * S_LEN;
    unsigned short* P        = ldsP[wave];

    // Q A-fragments for the whole key loop (K=64 -> 2 fragments)
    const unsigned short* qrow = Qb + (size_t)(q0 + ln) * DK;
    v16bf qa0 = load_frag(qrow + 0 + khalf);
    v16bf qa1 = load_frag(qrow + 32 + khalf);

    float m_run[8], l_run[8];
    v8f acc[4] = {};
#pragma unroll
    for (int r = 0; r < 8; ++r) { m_run[r] = -3.0e38f; l_run[r] = 0.0f; }

    for (int c0 = 0; c0 < S_LEN; c0 += 32) {
        // ---- scores: two 16x16 tiles over K=64 ----
        v8f s0 = {}, s1 = {};
        {
            const unsigned short* kr0 = Kb + (size_t)(c0 + ln) * DK;
            const unsigned short* kr1 = Kb + (size_t)(c0 + 16 + ln) * DK;
            s0 = wmma_bf16(qa0, load_frag(kr0 + khalf), s0);
            s0 = wmma_bf16(qa1, load_frag(kr0 + 32 + khalf), s0);
            s1 = wmma_bf16(qa0, load_frag(kr1 + khalf), s1);
            s1 = wmma_bf16(qa1, load_frag(kr1 + 32 + khalf), s1);
        }

        // ---- scale, mask, online softmax (row reductions over 16-lane groups) ----
#pragma unroll
        for (int r = 0; r < 8; ++r) {
            int qrowi = q0 + mbase + r;
            int j0 = c0 + ln, j1 = c0 + 16 + ln;
            float v0 = s0[r] * 0.125f;
            float v1 = s1[r] * 0.125f;
            if (mb[(size_t)qrowi * S_LEN + j0] == 0) v0 = -1.0e9f;
            if (mb[(size_t)qrowi * S_LEN + j1] == 0) v1 = -1.0e9f;

            float mx = fmaxf(v0, v1);
            mx = fmaxf(mx, __shfl_xor(mx, 1, 32));
            mx = fmaxf(mx, __shfl_xor(mx, 2, 32));
            mx = fmaxf(mx, __shfl_xor(mx, 4, 32));
            mx = fmaxf(mx, __shfl_xor(mx, 8, 32));

            float mnew = fmaxf(m_run[r], mx);
            float corr = __expf(m_run[r] - mnew);
            float e0 = __expf(v0 - mnew);
            float e1 = __expf(v1 - mnew);
            float sum = e0 + e1;
            sum += __shfl_xor(sum, 1, 32);
            sum += __shfl_xor(sum, 2, 32);
            sum += __shfl_xor(sum, 4, 32);
            sum += __shfl_xor(sum, 8, 32);

            l_run[r] = l_run[r] * corr + sum;
            m_run[r] = mnew;
            acc[0][r] *= corr; acc[1][r] *= corr;
            acc[2][r] *= corr; acc[3][r] *= corr;

            // stage P (C-layout -> row-major LDS [16][32])
            int prow = mbase + r;
            P[prow * 32 + ln]      = f32_to_bf16_rne(e0);
            P[prow * 32 + 16 + ln] = f32_to_bf16_rne(e1);
        }

        // ---- PV: A = P from LDS, B = Vt rows (contiguous keys) ----
        v16bf pa = load_frag(P + ln * 32 + khalf);
#pragma unroll
        for (int t = 0; t < 4; ++t) {
            const unsigned short* vrow = Vb + (size_t)(t * 16 + ln) * S_LEN + c0;
            acc[t] = wmma_bf16(pa, load_frag(vrow + khalf), acc[t]);
        }
    }

    // ---- epilogue: divide by l, write bf16 context [B][S][D] ----
#pragma unroll
    for (int r = 0; r < 8; ++r) {
        int srow  = q0 + mbase + r;
        float inv = 1.0f / l_run[r];
#pragma unroll
        for (int t = 0; t < 4; ++t) {
            int d = t * 16 + ln;
            ctx[((size_t)(b * S_LEN + srow)) * D_MODEL + h * DK + d] =
                f32_to_bf16_rne(acc[t][r] * inv);
        }
    }
}

// ---------------- launch ----------------
extern "C" void kernel_launch(void* const* d_in, const int* in_sizes, int n_in,
                              void* d_out, int out_size, void* d_ws, size_t ws_size,
                              hipStream_t stream) {
    const float* query = (const float*)d_in[0];
    const float* key   = (const float*)d_in[1];
    const float* value = (const float*)d_in[2];
    const int*   maskp = (const int*)d_in[3];
    const float* Wq = (const float*)d_in[4];
    const float* bq = (const float*)d_in[5];
    const float* Wk = (const float*)d_in[6];
    const float* bk = (const float*)d_in[7];
    const float* Wv = (const float*)d_in[8];
    const float* bv = (const float*)d_in[9];
    const float* Wo = (const float*)d_in[10];
    const float* bo = (const float*)d_in[11];

    const size_t NX = (size_t)4096 * 1024;  // activations  (8 MB as bf16)
    const size_t NW = (size_t)1024 * 1024;  // weights      (2 MB as bf16)
    unsigned short* ws  = (unsigned short*)d_ws;
    unsigned short* qbf = ws;
    unsigned short* kbf = qbf + NX;
    unsigned short* vbf = kbf + NX;
    unsigned short* wq  = vbf + NX;
    unsigned short* wk  = wq + NW;
    unsigned short* wv  = wk + NW;
    unsigned short* wo  = wv + NW;
    unsigned short* Qh  = wo + NW;
    unsigned short* Kh  = Qh + NX;
    unsigned short* Vt  = Kh + NX;
    unsigned short* ctx = Vt + NX;

    // 1) convert everything to bf16
    int nx4 = (int)(NX / 4), nw4 = (int)(NW / 4);
    cvt4_kernel<<<(nx4 + 255) / 256, 256, 0, stream>>>(query, qbf, nx4);
    cvt4_kernel<<<(nx4 + 255) / 256, 256, 0, stream>>>(key,   kbf, nx4);
    cvt4_kernel<<<(nx4 + 255) / 256, 256, 0, stream>>>(value, vbf, nx4);
    cvt4_kernel<<<(nw4 + 255) / 256, 256, 0, stream>>>(Wq, wq, nw4);
    cvt4_kernel<<<(nw4 + 255) / 256, 256, 0, stream>>>(Wk, wk, nw4);
    cvt4_kernel<<<(nw4 + 255) / 256, 256, 0, stream>>>(Wv, wv, nw4);
    cvt4_kernel<<<(nw4 + 255) / 256, 256, 0, stream>>>(Wo, wo, nw4);

    // 2) Q/K/V projections (LDS-tiled WMMA bf16, async-copy pipelined)
    dim3 ggrid(8, 32), gblk(256);
    gemm_bf16_kernel<0><<<ggrid, gblk, 0, stream>>>(qbf, wq, bq, (void*)Qh);
    gemm_bf16_kernel<0><<<ggrid, gblk, 0, stream>>>(kbf, wk, bk, (void*)Kh);
    gemm_bf16_kernel<1><<<ggrid, gblk, 0, stream>>>(vbf, wv, bv, (void*)Vt);

    // 3) flash attention (WMMA for QK^T and PV)
    attn_kernel<<<512, 256, 0, stream>>>(Qh, Kh, Vt, maskp, ctx);

    // 4) output projection -> f32 d_out
    gemm_bf16_kernel<2><<<ggrid, gblk, 0, stream>>>(ctx, wo, bo, d_out);
}
